// svsLanguage_46978352283806
// MI455X (gfx1250) — matrix operand order repaired and checked by reading, not compiled, not run on hardware
//
#include <hip/hip_runtime.h>
#include <cstring>

typedef unsigned short u16;
typedef __bf16 bf16t;
typedef bf16t v16bf __attribute__((ext_vector_type(16)));
typedef float v8f __attribute__((ext_vector_type(8)));

union Frag { v16bf v; uint4 q[2]; };

__device__ inline u16 f2bf(float f) {
  union { float f; unsigned u; } c; c.f = f;
  unsigned u = c.u;
  if ((u & 0x7F800000u) == 0x7F800000u) return (u16)(u >> 16);   // inf/nan
  return (u16)((u + 0x7FFFu + ((u >> 16) & 1u)) >> 16);          // RNE
}
// branch-free activations: v_exp_f32 + v_rcp_f32, no divergent ocml paths
__device__ inline float sigf(float x) {
  return __builtin_amdgcn_rcpf(1.f + __expf(-x));
}
__device__ inline float tanh_fast(float x) {
  float t = __expf(-2.f * fabsf(x));              // always <= 1, no overflow
  float r = (1.f - t) * __builtin_amdgcn_rcpf(1.f + t);
  return copysignf(r, x);
}

// ---------------------------------------------------------------------------
// Generic fused WMMA body.
//   GATES==4 : LSTM cell.  gates laid out [i|f|g|o] each padded to Hp rows.
//              C = sig(f)*c + sig(i)*tanh(g);  h = sig(o)*tanh(C)
//   GATES==1 : plain linear (+bias, optional relu)
// A inputs are bf16 row-major [128, Kp] (Kp multiple of 32, zero padded).
// Weights are bf16 packed [GATES*Hp, Kp], gate/row/col zero padded.
// One workgroup = 8 wave32 = all 8 M-tiles of B=128; blockIdx.x = N tile.
// ---------------------------------------------------------------------------
struct GemmArgs {
  const u16* A0; const u16* W0; int K0;
  const u16* A1; const u16* W1; int K1;     // optional second input (A1 may be null)
  const float* bias;                         // [GATES*Hp]
  int Hp;                                    // padded per-gate width (mult of 16)
  int Nreal;                                 // valid columns for Of/Obf writes
  float* cstate;                             // [128,Hp] f32 (LSTM)
  u16*   hstate;                             // [128,Hp] bf16 (LSTM, full write)
  float* Yf;                                 // plain: full padded f32 out [128,Hp]
  float* Of;  int Of_stride;                 // guarded f32 out (ncol<Nreal)
  u16*   Obf; int Obf_rs; int Obf_cs;        // guarded bf16 out, custom strides
  int relu;                                  // relu on Of/Obf (and Yf in plain mode)
};

template <int GATES>
__device__ inline void rnn_body(const GemmArgs& g) {
  const int ntile = blockIdx.x;
  if (ntile * 16 >= g.Hp) return;                 // allows mixed-Hp dual launches
  const int lane  = threadIdx.x & 31;
  const int wave  = threadIdx.x >> 5;
  const int m0    = wave * 16;                    // M = 128 fixed
  const int ncol  = ntile * 16 + (lane & 15);

  v8f acc[GATES];
#pragma unroll
  for (int q = 0; q < GATES; ++q) {
    float b = g.bias ? g.bias[q * g.Hp + ncol] : 0.f;
#pragma unroll
    for (int e = 0; e < 8; ++e) acc[q][e] = b;
  }

  const int mrow = m0 + (lane & 15);
  const int aoff = (lane >> 4) << 3;   // A: K chunks at {aoff, 16+aoff}
  const int boff = (lane >> 4) << 3;   // B: K chunks at {boff, 16+boff}

#pragma unroll 1
  for (int s = 0; s < 2; ++s) {
    const u16* A = s ? g.A1 : g.A0;
    const u16* W = s ? g.W1 : g.W0;
    const int  K = s ? g.K1 : g.K0;
    if (!A) continue;
    const u16* arow = A + (size_t)mrow * K + aoff;
    const u16* brow = W + (size_t)(ntile * 16 + (lane & 15)) * K + boff;
    const size_t gstride = (size_t)g.Hp * K;
#pragma unroll 1
    for (int k0 = 0; k0 < K; k0 += 32) {
      // issue ALL loads first so WMMAs can start on partial loadcnt waits
      Frag af;
      af.q[0] = *(const uint4*)(arow + k0);
      af.q[1] = *(const uint4*)(arow + k0 + 16);
      Frag bf[GATES];
#pragma unroll
      for (int q = 0; q < GATES; ++q) {
        const u16* bp = brow + (size_t)q * gstride + k0;
        bf[q].q[0] = *(const uint4*)(bp);
        bf[q].q[1] = *(const uint4*)(bp + 16);
      }
#pragma unroll
      for (int q = 0; q < GATES; ++q)
        acc[q] = __builtin_amdgcn_wmma_f32_16x16x32_bf16(
            false, af.v, false, bf[q].v, (short)0, acc[q], false, false);
    }
  }

  const int mbase = m0 + ((lane >> 4) << 3);      // D elem e -> row mbase+e
  if (GATES == 4) {
#pragma unroll
    for (int e = 0; e < 8; ++e) {
      const int m = mbase + e;
      const size_t si = (size_t)m * g.Hp + ncol;
      float iv = sigf(acc[0][e]);
      float fv = sigf(acc[1][e]);
      float gv = tanh_fast(acc[2][e]);
      float ov = sigf(acc[3 % GATES][e]);
      float c  = fv * g.cstate[si] + iv * gv;
      g.cstate[si] = c;
      float h = ov * tanh_fast(c);
      g.hstate[si] = f2bf(h);
      if (ncol < g.Nreal) {
        float ho = g.relu ? fmaxf(h, 0.f) : h;
        if (g.Of)  g.Of[(size_t)m * g.Of_stride + ncol] = ho;
        if (g.Obf) g.Obf[(size_t)m * g.Obf_rs + (size_t)ncol * g.Obf_cs] = f2bf(ho);
      }
    }
  } else {
#pragma unroll
    for (int e = 0; e < 8; ++e) {
      const int m = mbase + e;
      float y = acc[0][e];
      if (g.relu) y = fmaxf(y, 0.f);
      if (g.Yf) g.Yf[(size_t)m * g.Hp + ncol] = y;
      if (ncol < g.Nreal) {
        if (g.Of)  g.Of[(size_t)m * g.Of_stride + ncol] = y;
        if (g.Obf) g.Obf[(size_t)m * g.Obf_rs + (size_t)ncol * g.Obf_cs] = f2bf(y);
      }
    }
  }
}

template <int GATES>
__global__ __launch_bounds__(256) void wmma_rnn(GemmArgs g) { rnn_body<GATES>(g); }

// two independent problems in one launch (gridDim.y == 2) — removes one
// kernel latency from the serial per-timestep chain
template <int GATES>
__global__ __launch_bounds__(256) void wmma_rnn2(GemmArgs a, GemmArgs b) {
  rnn_body<GATES>(blockIdx.y ? b : a);
}

// ------------------------- utility kernels --------------------------------
__global__ void pack_w_kernel(const float* src, u16* dst,
                              int N, int K, int Kp, int Hp, int gates) {
  size_t i = (size_t)blockIdx.x * blockDim.x + threadIdx.x;
  size_t tot = (size_t)gates * Hp * Kp;
  if (i >= tot) return;
  int k = (int)(i % Kp);
  int r = (int)(i / Kp);
  int gq = r / Hp, p = r % Hp;
  float v = 0.f;
  if (p < N && k < K) v = src[(size_t)(gq * N + p) * K + k];
  dst[i] = f2bf(v);
}

__global__ void pack_bias_kernel(const float* b0, const float* b1, float* dst,
                                 int N, int Hp, int gates) {
  int i = blockIdx.x * blockDim.x + threadIdx.x;
  if (i >= gates * Hp) return;
  int gq = i / Hp, p = i % Hp;
  float v = 0.f;
  if (p < N) { v = b0[gq * N + p]; if (b1) v += b1[gq * N + p]; }
  dst[i] = v;
}

__global__ void cvt_bf16_kernel(const float* s, u16* d, size_t n) {
  size_t i = (size_t)blockIdx.x * blockDim.x + threadIdx.x;
  if (i < n) d[i] = f2bf(s[i]);
}

__global__ void zero_kernel(unsigned* p, size_t n) {
  size_t i = (size_t)blockIdx.x * blockDim.x + threadIdx.x;
  if (i < n) p[i] = 0u;
}

__global__ void reparam_kernel(const float* muf, const float* lvf,
                               const float* eps, u16* z, int Hp, int Nreal) {
  int b = blockIdx.x, n = threadIdx.x;
  float v = 0.f;
  if (n < Nreal)
    v = eps[b * Nreal + n] * __expf(0.5f * lvf[(size_t)b * Hp + n])
        + muf[(size_t)b * Hp + n];
  if (n < Hp) z[(size_t)b * Hp + n] = f2bf(v);
}

// ------------------------------ host side ---------------------------------
namespace {
enum { SEQv=512, Bv=128, INv=128, OUTv=512, MID_Iv=60, MID_Ov=800,
       LATv=20, V_MIDv=800, V_LATv=50 };
inline size_t al256(size_t x) { return (x + 255) & ~(size_t)255; }
}

extern "C" void kernel_launch(void* const* d_in, const int* in_sizes, int n_in,
                              void* d_out, int out_size, void* d_ws, size_t ws_size,
                              hipStream_t stream) {
  (void)in_sizes; (void)n_in; (void)out_size; (void)ws_size;
  auto in = [&](int i) { return (const float*)d_in[i]; };
  // input order: x, eps, then params in dict order, each lstm = {Wih,Whh,bih,bhh},
  // each linear = {W,b}
  const float* x_f   = in(0);
  const float* eps_f = in(1);
  const int FC1=2, FC21=6, FC22=10, L3=14, L4=18, L5=22, L6=26,
            LIN1=30, LIN2=32, IDL=34, LD1=36, LD2=38;

  // ---- workspace bump allocator ----
  char* ws = (char*)d_ws; size_t off = 0;
  auto alloc = [&](size_t bytes) { void* p = ws + off; off = al256(off + bytes); return p; };
  auto aU = [&](size_t n) { return (u16*)alloc(n * 2); };
  auto aF = [&](size_t n) { return (float*)alloc(n * 4); };

  u16* xbf = aU((size_t)SEQv * Bv * INv);
  // packed weights (bf16) + biases (f32)
  u16 *pFc1ih = aU(4*64*128), *pFc1hh = aU(4*64*64);   float* bFc1 = aF(256);
  u16 *p21ih  = aU(4*32*64),  *p21hh  = aU(4*32*32);   float* b21  = aF(128);
  u16 *p22ih  = aU(4*32*64),  *p22hh  = aU(4*32*32);   float* b22  = aF(128);
  u16 *p3ih   = aU(4*64*32),  *p3hh   = aU(4*64*64);   float* b3   = aF(256);
  u16 *p4ih   = aU((size_t)4*800*64),  *p4hh = aU((size_t)4*800*800); float* b4 = aF(3200);
  u16 *p5ih   = aU((size_t)4*512*800), *p5hh = aU((size_t)4*512*512); float* b5 = aF(2048);
  u16 *p6ih   = aU((size_t)4*512*800), *p6hh = aU((size_t)4*512*512); float* b6 = aF(2048);
  u16 *pLin1  = aU((size_t)800*10240);                 float* bLin1 = aF(800);
  u16 *pLin2  = aU(64*800);                            float* bLin2 = aF(64);
  u16 *pId    = aU(64*64);                             float* bId   = aF(64);
  u16 *pLd1   = aU(32*64);                             float* bLd1  = aF(32);
  u16 *pLd2   = aU(32*512);                            float* bLd2  = aF(32);
  // activations / states
  u16* h1seq  = aU((size_t)SEQv * Bv * 64);            // relu'd fc1 outputs
  u16* flat21 = aU((size_t)Bv * LATv * SEQv);
  u16* flat22 = aU((size_t)Bv * LATv * SEQv);
  u16* fc1_h[2] = { aU(Bv*64), aU(Bv*64) };            float* fc1_c = aF(Bv*64);
  u16* h21_h[2] = { aU(Bv*32), aU(Bv*32) };            float* h21_c = aF(Bv*32);
  u16* h22_h[2] = { aU(Bv*32), aU(Bv*32) };            float* h22_c = aF(Bv*32);
  u16* vbufA  = aU(Bv*800);
  u16* vbufB  = aU(Bv*800);
  float* muf  = aF(Bv*64);  float* lvf = aF(Bv*64);
  u16* d3_h[2] = { aU(Bv*64),  aU(Bv*64)  };           float* d3_c = aF(Bv*64);
  u16* d4_h[2] = { aU(Bv*800), aU(Bv*800) };           float* d4_c = aF(Bv*800);
  u16* d5_h[2] = { aU(Bv*512), aU(Bv*512) };           float* d5_c = aF(Bv*512);
  u16* d6_h[2] = { aU(Bv*512), aU(Bv*512) };           float* d6_c = aF(Bv*512);
  u16* obuf   = aU(Bv*32);

  // ---- helpers ----
  auto packW = [&](int idx, u16* dst, int N, int K, int Kp, int Hp, int gates) {
    size_t tot = (size_t)gates * Hp * Kp;
    pack_w_kernel<<<dim3((unsigned)((tot + 255) / 256)), 256, 0, stream>>>(
        in(idx), dst, N, K, Kp, Hp, gates);
  };
  auto packB = [&](const float* b0, const float* b1, float* dst, int N, int Hp, int gates) {
    int tot = gates * Hp;
    pack_bias_kernel<<<dim3((tot + 255) / 256), 256, 0, stream>>>(b0, b1, dst, N, Hp, gates);
  };
  auto zero = [&](void* p, size_t bytes) {
    size_t n = bytes / 4;
    zero_kernel<<<dim3((unsigned)((n + 255) / 256)), 256, 0, stream>>>((unsigned*)p, n);
  };
  auto gemm = [&](int gates, const GemmArgs& a) {
    int ntiles = a.Hp / 16;
    if (gates == 4) wmma_rnn<4><<<dim3(ntiles), 256, 0, stream>>>(a);
    else            wmma_rnn<1><<<dim3(ntiles), 256, 0, stream>>>(a);
  };
  auto gemm2 = [&](int gates, const GemmArgs& a, const GemmArgs& b) {
    int nt = a.Hp > b.Hp ? a.Hp / 16 : b.Hp / 16;
    if (gates == 4) wmma_rnn2<4><<<dim3(nt, 2), 256, 0, stream>>>(a, b);
    else            wmma_rnn2<1><<<dim3(nt, 2), 256, 0, stream>>>(a, b);
  };

  // ---- 0) convert x, pack all weights ----
  {
    size_t n = (size_t)SEQv * Bv * INv;
    cvt_bf16_kernel<<<dim3((unsigned)((n + 255) / 256)), 256, 0, stream>>>(x_f, xbf, n);
  }
  packW(FC1+0, pFc1ih, 60, 128, 128, 64, 4);  packW(FC1+1, pFc1hh, 60, 60, 64, 64, 4);
  packB(in(FC1+2), in(FC1+3), bFc1, 60, 64, 4);
  packW(FC21+0, p21ih, 20, 60, 64, 32, 4);    packW(FC21+1, p21hh, 20, 20, 32, 32, 4);
  packB(in(FC21+2), in(FC21+3), b21, 20, 32, 4);
  packW(FC22+0, p22ih, 20, 60, 64, 32, 4);    packW(FC22+1, p22hh, 20, 20, 32, 32, 4);
  packB(in(FC22+2), in(FC22+3), b22, 20, 32, 4);
  packW(L3+0, p3ih, 50, 20, 32, 64, 4);       packW(L3+1, p3hh, 50, 50, 64, 64, 4);
  packB(in(L3+2), in(L3+3), b3, 50, 64, 4);
  packW(L4+0, p4ih, 800, 50, 64, 800, 4);     packW(L4+1, p4hh, 800, 800, 800, 800, 4);
  packB(in(L4+2), in(L4+3), b4, 800, 800, 4);
  packW(L5+0, p5ih, 512, 800, 800, 512, 4);   packW(L5+1, p5hh, 512, 512, 512, 512, 4);
  packB(in(L5+2), in(L5+3), b5, 512, 512, 4);
  packW(L6+0, p6ih, 512, 800, 800, 512, 4);   packW(L6+1, p6hh, 512, 512, 512, 512, 4);
  packB(in(L6+2), in(L6+3), b6, 512, 512, 4);
  packW(LIN1, pLin1, 800, 10240, 10240, 800, 1); packB(in(LIN1+1), nullptr, bLin1, 800, 800, 1);
  packW(LIN2, pLin2, 50, 800, 800, 64, 1);       packB(in(LIN2+1), nullptr, bLin2, 50, 64, 1);
  packW(IDL,  pId, 50, 50, 64, 64, 1);           packB(in(IDL+1),  nullptr, bId, 50, 64, 1);
  packW(LD1,  pLd1, 20, 50, 64, 32, 1);          packB(in(LD1+1),  nullptr, bLd1, 20, 32, 1);
  packW(LD2,  pLd2, 20, 512, 512, 32, 1);        packB(in(LD2+1),  nullptr, bLd2, 20, 32, 1);

  // ---- 1) zero-init recurrent states ----
  zero(fc1_h[0], Bv*64*2); zero(fc1_c, Bv*64*4);
  zero(h21_h[0], Bv*32*2); zero(h21_c, Bv*32*4);
  zero(h22_h[0], Bv*32*2); zero(h22_c, Bv*32*4);
  zero(d4_h[0], (size_t)Bv*800*2); zero(d4_c, (size_t)Bv*800*4);
  zero(d5_h[0], (size_t)Bv*512*2); zero(d5_c, (size_t)Bv*512*4);
  zero(d6_h[0], (size_t)Bv*512*2); zero(d6_c, (size_t)Bv*512*4);

  // ---- 2) encoder fc1 LSTM over 512 steps (relu'd seq -> h1seq) ----
  int c1 = 0;
  for (int t = 0; t < SEQv; ++t) {
    GemmArgs a{};
    a.A0 = fc1_h[c1]; a.W0 = pFc1hh; a.K0 = 64;
    a.A1 = xbf + (size_t)t * Bv * INv; a.W1 = pFc1ih; a.K1 = 128;
    a.bias = bFc1; a.Hp = 64; a.Nreal = 64;
    a.cstate = fc1_c; a.hstate = fc1_h[1 - c1];
    a.Obf = h1seq + (size_t)t * Bv * 64; a.Obf_rs = 64; a.Obf_cs = 1; a.relu = 1;
    gemm(4, a); c1 ^= 1;
  }

  // ---- 3) both head LSTMs per step fused in one launch, latent-major out ----
  int c21 = 0, c22 = 0;
  for (int t = 0; t < SEQv; ++t) {
    GemmArgs a{};
    a.A0 = h21_h[c21]; a.W0 = p21hh; a.K0 = 32;
    a.A1 = h1seq + (size_t)t * Bv * 64; a.W1 = p21ih; a.K1 = 64;
    a.bias = b21; a.Hp = 32; a.Nreal = LATv;
    a.cstate = h21_c; a.hstate = h21_h[1 - c21];
    a.Obf = flat21 + t; a.Obf_rs = LATv * SEQv; a.Obf_cs = SEQv; a.relu = 0;

    GemmArgs b{};
    b.A0 = h22_h[c22]; b.W0 = p22hh; b.K0 = 32;
    b.A1 = h1seq + (size_t)t * Bv * 64; b.W1 = p22ih; b.K1 = 64;
    b.bias = b22; b.Hp = 32; b.Nreal = LATv;
    b.cstate = h22_c; b.hstate = h22_h[1 - c22];
    b.Obf = flat22 + t; b.Obf_rs = LATv * SEQv; b.Obf_cs = SEQv; b.relu = 0;

    gemm2(4, a, b); c21 ^= 1; c22 ^= 1;
  }

  // ---- 4) lin1/lin2 for both heads (dual launches) -> mu, logvar ----
  float* out_f = (float*)d_out;
  const size_t MU_OFF = (size_t)2 * SEQv * Bv * OUTv;
  const size_t LV_OFF = MU_OFF + (size_t)Bv * V_LATv;
  {
    GemmArgs a{}, b{};
    a.A0 = flat21; a.W0 = pLin1; a.K0 = LATv * SEQv;
    a.bias = bLin1; a.Hp = 800; a.Nreal = 800;
    a.Obf = vbufA; a.Obf_rs = 800; a.Obf_cs = 1; a.relu = 1;
    b = a; b.A0 = flat22; b.Obf = vbufB;
    gemm2(1, a, b);

    GemmArgs c{}, d{};
    c.A0 = vbufA; c.W0 = pLin2; c.K0 = 800;
    c.bias = bLin2; c.Hp = 64; c.Nreal = V_LATv;
    c.Yf = muf; c.Of = out_f + MU_OFF; c.Of_stride = V_LATv; c.relu = 1;
    d = c; d.A0 = vbufB; d.Yf = lvf; d.Of = out_f + LV_OFF;
    gemm2(1, c, d);
  }

  // ---- 5) reparameterize -> z (= lstm3 initial h); c = idlinear(z); o = ld1(z)
  reparam_kernel<<<dim3(Bv), dim3(64), 0, stream>>>(muf, lvf, eps_f, d3_h[0], 64, V_LATv);
  {
    GemmArgs a{}, b{};
    a.A0 = d3_h[0]; a.W0 = pId; a.K0 = 64;
    a.bias = bId; a.Hp = 64; a.Nreal = 64; a.Yf = d3_c; a.relu = 0;
    b.A0 = d3_h[0]; b.W0 = pLd1; b.K0 = 64;
    b.bias = bLd1; b.Hp = 32; b.Nreal = 32;
    b.Obf = obuf; b.Obf_rs = 32; b.Obf_cs = 1; b.relu = 0;
    gemm2(1, a, b);
  }

  // ---- 6) decoder: 512 steps of lstm3 -> lstm4 -> (lstm5 || lstm6) -> ld2 ----
  int c3 = 0, c4 = 0, c5 = 0, c6 = 0;
  for (int t = 0; t < SEQv; ++t) {
    GemmArgs a{};                                    // lstm3: (o, h3st) H=50/Hp=64
    a.A0 = obuf; a.W0 = p3ih; a.K0 = 32;
    a.A1 = d3_h[c3]; a.W1 = p3hh; a.K1 = 64;
    a.bias = b3; a.Hp = 64; a.Nreal = 64;
    a.cstate = d3_c; a.hstate = d3_h[1 - c3];
    gemm(4, a);

    GemmArgs b{};                                    // lstm4: (h, h2) H=800
    b.A0 = d3_h[1 - c3]; b.W0 = p4ih; b.K0 = 64;
    b.A1 = d4_h[c4]; b.W1 = p4hh; b.K1 = 800;
    b.bias = b4; b.Hp = 800; b.Nreal = 800;
    b.cstate = d4_c; b.hstate = d4_h[1 - c4];
    gemm(4, b);

    GemmArgs c{};                                    // lstm5: h3 -> muTheta[t]
    c.A0 = d4_h[1 - c4]; c.W0 = p5ih; c.K0 = 800;
    c.A1 = d5_h[c5]; c.W1 = p5hh; c.K1 = 512;
    c.bias = b5; c.Hp = 512; c.Nreal = 512;
    c.cstate = d5_c; c.hstate = d5_h[1 - c5];
    c.Of = out_f + (size_t)t * Bv * OUTv; c.Of_stride = OUTv; c.relu = 0;

    GemmArgs d{};                                    // lstm6: hv -> logvarTheta[t]
    d.A0 = d4_h[1 - c4]; d.W0 = p6ih; d.K0 = 800;
    d.A1 = d6_h[c6]; d.W1 = p6hh; d.K1 = 512;
    d.bias = b6; d.Hp = 512; d.Nreal = 512;
    d.cstate = d6_c; d.hstate = d6_h[1 - c6];
    d.Of = out_f + (size_t)SEQv * Bv * OUTv + (size_t)t * Bv * OUTv;
    d.Of_stride = OUTv; d.relu = 0;
    gemm2(4, c, d);                                  // run lstm5 & lstm6 concurrently

    GemmArgs e{};                                    // o = lineardown2(h3)
    e.A0 = d5_h[1 - c5]; e.W0 = pLd2; e.K0 = 512;
    e.bias = bLd2; e.Hp = 32; e.Nreal = 32;
    e.Obf = obuf; e.Obf_rs = 32; e.Obf_cs = 1; e.relu = 0;
    gemm(1, e);

    c3 ^= 1; c4 ^= 1; c5 ^= 1; c6 ^= 1;
  }
}
